// FNO_multi_73400991089327
// MI455X (gfx1250) — matrix operand order repaired and checked by reading, not compiled, not run on hardware
//
#include <hip/hip_runtime.h>
#include <hip/hip_bf16.h>

// ---------------------------------------------------------------------------
// FNO on MI455X (gfx1250, wave32). All channel mixes, truncated DFTs and the
// MLP head are mapped onto v_wmma_f32_16x16x32_f16 (f16 operands, f32 accum).
// Strided-gather operands are staged through LDS with the Tensor Data Mover
// (tensor_load_to_lds + s_wait_tensorcnt). Activations are f16 [point][32ch];
// hot set ~35MB => resident in the 192MB L2.
// ---------------------------------------------------------------------------

#define BB   16
#define VV   3
#define SX   106
#define SY   106
#define TIN  10
#define STEP 5
#define M1   16
#define M2   16
#define WCH  32
#define NL   6

#define NPT    (BB * VV * SX * SY)     // 539328 points
#define NTILES (NPT / 16)              // 33708 tiles of 16 points
#define SP2    (SX * SY)               // 11236

typedef __attribute__((ext_vector_type(16))) _Float16 v16h;
typedef __attribute__((ext_vector_type(8)))  float    v8f;

typedef __attribute__((ext_vector_type(4))) unsigned int tdm_g0_t;
typedef __attribute__((ext_vector_type(8))) int          tdm_g1_t;
typedef __attribute__((ext_vector_type(4))) int          tdm_g23_t;
typedef __attribute__((ext_vector_type(8))) int          tdm_g4_t;

// Tensor Data Mover: DMA a [nrows x row_elems] f16 tile (row stride in elems)
// from global memory into LDS at byte offset lds_off. Descriptor bitfields per
// CDNA5 ISA (count=1, type=2 "image", data_size=2B).
// clang-23 toolchain: 6-arg form (g0, g1, g2, g3, extra, cpol).
__device__ __forceinline__ void tdm_load_f16(unsigned lds_off, const void* gsrc,
                                             unsigned row_elems, unsigned nrows,
                                             unsigned row_stride_elems) {
  unsigned long long ga = (unsigned long long)(uintptr_t)gsrc;
  tdm_g0_t g0;
  g0[0] = 1u;                                              // count=1 (valid)
  g0[1] = lds_off;                                         // lds_addr (bytes)
  g0[2] = (unsigned)ga;                                    // global_addr[31:0]
  g0[3] = (unsigned)((ga >> 32) & 0x01FFFFFFull) | (2u << 30); // addr[56:32]|type=2
  tdm_g1_t g1;
  g1[0] = (int)(1u << 16);                                 // data_size=1 -> 2 bytes
  g1[1] = (int)(row_elems << 16);                          // tensor_dim0[15:0]
  g1[2] = (int)((row_elems >> 16) | (nrows << 16));        // td0[31:16] | td1[15:0]
  g1[3] = (int)(((nrows >> 16) & 0xFFFFu) | ((row_elems & 0xFFFFu) << 16)); // td1 hi | tile_dim0
  g1[4] = (int)(nrows & 0xFFFFu);                          // tile_dim1
  g1[5] = (int)row_stride_elems;                           // tensor_dim0_stride lo32
  g1[6] = 0;
  g1[7] = 0;
  tdm_g23_t gz = {0, 0, 0, 0};
  tdm_g4_t  g4 = {0, 0, 0, 0, 0, 0, 0, 0};
  __builtin_amdgcn_tensor_load_to_lds(g0, g1, gz, gz, g4, 0);
}

// The TDM write to LDS is invisible to the optimizer (it only sees an integer
// LDS offset), so a never-"stored" LDS array would have all its loads folded
// to undef. Publish the DMA result with an undeletable volatile self-write of
// each thread's own slot: creates a real store to the global (dynamic index)
// without changing the data.
#define TDM_PUBLISH(arr)                                        \
  do {                                                          \
    volatile _Float16* _vp = (arr);                             \
    _vp[threadIdx.x] = _vp[threadIdx.x];                        \
  } while (0)

__device__ __forceinline__ v8f wmma32(v16h a, v16h b, v8f c) {
  return __builtin_amdgcn_wmma_f32_16x16x32_f16(false, a, false, b, (short)0, c,
                                                false, false);
}

__device__ __forceinline__ float gelu_f(float v) {
  return 0.5f * v * (1.0f + erff(v * 0.70710678118654752440f));
}

// A fragment (16x32, f16) from row-major [m][k], k contiguous, leading dim ldk
__device__ __forceinline__ v16h load_a_rm(const _Float16* __restrict__ base, int ldk) {
  int l = threadIdx.x & 31;
  int m = l & 15, kh = (l >> 4) * 8;
  const _Float16* r = base + (size_t)m * ldk;
  v16h a;
#pragma unroll
  for (int i = 0; i < 8; ++i) { a[i] = r[kh + i]; a[8 + i] = r[16 + kh + i]; }
  return a;
}

// B fragment (32x16, f16) from transposed table [n][k] (k contiguous, ld=ldk)
__device__ __forceinline__ v16h load_b_t(const _Float16* bt, int ldk) {
  int l = threadIdx.x & 31;
  int n = l & 15, kb = (l >> 4) * 8;
  const _Float16* r = bt + (size_t)n * ldk + kb;
  v16h b;
#pragma unroll
  for (int i = 0; i < 8; ++i) { b[i] = r[i]; b[8 + i] = r[16 + i]; }
  return b;
}

// ------------------------- prep kernels ------------------------------------

__global__ void k_prep_weights(const float* __restrict__ fc0w,
                               const float* __restrict__ m1w,
                               const float* __restrict__ m2w,
                               const float* __restrict__ www,
                               const float* __restrict__ fc1w,
                               const float* __restrict__ fc2w,
                               _Float16* __restrict__ FC0T,
                               _Float16* __restrict__ M1T,
                               _Float16* __restrict__ M2T,
                               _Float16* __restrict__ WWT,
                               _Float16* __restrict__ FC1T,
                               _Float16* __restrict__ FC2T) {
  int id = blockIdx.x * blockDim.x + threadIdx.x;
  if (id < 1024) {
    int o = id >> 5, k = id & 31;
    FC0T[id] = (_Float16)((k < TIN + 2) ? fc0w[o * (TIN + 2) + k] : 0.0f);
    return;
  }
  id -= 1024;
  if (id < NL * 1024) { M1T[id] = (_Float16)m1w[id]; return; }
  id -= NL * 1024;
  if (id < NL * 1024) { M2T[id] = (_Float16)m2w[id]; return; }
  id -= NL * 1024;
  if (id < NL * 1024) { WWT[id] = (_Float16)www[id]; return; }
  id -= NL * 1024;
  if (id < 256 * 32) { FC1T[id] = (_Float16)fc1w[id]; return; }
  id -= 256 * 32;
  if (id < 16 * 256) {
    int o = id >> 8, k = id & 255;
    FC2T[id] = (_Float16)((o < STEP) ? fc2w[o * 256 + k] : 0.0f);
    return;
  }
}

__global__ void k_prep_tables(_Float16* __restrict__ FYC, _Float16* __restrict__ FYS,
                              _Float16* __restrict__ FXR, _Float16* __restrict__ FXI,
                              _Float16* __restrict__ IXR, _Float16* __restrict__ IXI,
                              _Float16* __restrict__ IY) {
  const float w0 = 6.28318530717958647692f / 106.0f;
  int id = blockIdx.x * blockDim.x + threadIdx.x;
  if (id < 16 * 128) {                 // forward-y: exp(-i w0 ky y)
    int ky = id >> 7, y = id & 127;
    float c = 0.f, s = 0.f;
    if (y < SY) { float t = w0 * (float)ky * (float)y; c = cosf(t); s = sinf(t); }
    FYC[id] = (_Float16)c;  FYS[id] = (_Float16)(-s);
    return;
  }
  id -= 16 * 128;
  if (id < 32 * 256) {                 // forward-x, K-folded [cos|sin] / [-sin|cos]
    int kx = id >> 8, k = id & 255;
    int x = k & 127; bool hi = (k >= 128);
    int kxp = (kx < 16) ? kx : (74 + kx);
    float c = 0.f, s = 0.f;
    if (x < SX) { float t = w0 * (float)kxp * (float)x; c = cosf(t); s = sinf(t); }
    FXR[id] = (_Float16)(hi ? s : c);
    FXI[id] = (_Float16)(hi ? c : -s);
    return;
  }
  id -= 32 * 256;
  if (id < 112 * 64) {                 // inverse-x: exp(+i w0 kxp x)
    int x = id >> 6, k = id & 63;
    int kx = k & 31; bool hi = (k >= 32);
    int kxp = (kx < 16) ? kx : (74 + kx);
    float c = 0.f, s = 0.f;
    if (x < SX) { float t = w0 * (float)kxp * (float)x; c = cosf(t); s = sinf(t); }
    IXR[id] = (_Float16)(hi ? -s : c);
    IXI[id] = (_Float16)(hi ? c : s);
    return;
  }
  id -= 112 * 64;
  if (id < 112 * 32) {                 // inverse-y to real, scale + conj doubling
    int y = id >> 5, k = id & 31;
    int ky = k & 15; bool hi = (k >= 16);
    float v = 0.f;
    if (y < SY) {
      float f = ((ky == 0) ? 1.0f : 2.0f) / (float)SP2;
      float t = w0 * (float)ky * (float)y;
      v = hi ? (-f * sinf(t)) : (f * cosf(t));
    }
    IY[id] = (_Float16)v;
    return;
  }
}

__global__ void k_gridterms(const float* __restrict__ bw, const float* __restrict__ bb,
                            const float* __restrict__ wb, const float* __restrict__ m2b,
                            _Float16* __restrict__ gt) {
  int id = blockIdx.x * blockDim.x + threadIdx.x;
  if (id >= NL * SP2 * WCH) return;
  int ch = id & 31;
  int sp = (id >> 5) % SP2;
  int i  = id / (SP2 * WCH);
  int yy = sp % SY, xx = sp / SY;
  float gx = 9.5f + (float)xx * (1.0f / 105.0f);
  float gy = -0.5f + (float)yy * (1.0f / 105.0f);
  int ic = i * WCH + ch;
  float v = bw[ic * 2 + 0] * gx + bw[ic * 2 + 1] * gy + bb[ic] + wb[ic] + m2b[ic];
  gt[id] = (_Float16)v;
}

// ------------------------- lift (fc0) --------------------------------------
__global__ void k_lift(const float* __restrict__ xin, const _Float16* __restrict__ FC0T,
                       const float* __restrict__ fc0b, _Float16* __restrict__ h) {
  int tile = blockIdx.x;
  int l = threadIdx.x;
  int m = l & 15, kh = (l >> 4) * 8;
  size_t p = (size_t)tile * 16 + m;
  int yy = (int)(p % SY), xx = (int)((p / SY) % SX);
  float gx = 9.5f + (float)xx * (1.0f / 105.0f);
  float gy = -0.5f + (float)yy * (1.0f / 105.0f);
  const float* feat = xin + p * TIN;
  v16h a;
#pragma unroll
  for (int i = 0; i < 8; ++i) {
    int k0 = kh + i;
    float f0 = (k0 < TIN) ? feat[k0] : (k0 == 10 ? gx : (k0 == 11 ? gy : 0.0f));
    a[i] = (_Float16)f0;
    a[8 + i] = (_Float16)0.0f;        // K 16..31 always zero padding
  }
#pragma unroll
  for (int nt = 0; nt < 2; ++nt) {
    v16h b = load_b_t(FC0T + nt * 16 * 32, 32);
    v8f acc = {};
    acc = wmma32(a, b, acc);
    int ch = nt * 16 + (l & 15), mb = (l >> 4) * 8;
    float bias = fc0b[ch];
#pragma unroll
    for (int j = 0; j < 8; ++j)
      h[((size_t)tile * 16 + mb + j) * WCH + ch] = (_Float16)(acc[j] + bias);
  }
}

// ------------------------- spectral conv -----------------------------------

// Forward DFT over y; h row (SY*WCH contiguous f16) staged to LDS via TDM.
__global__ void k_dfty(const _Float16* __restrict__ h, const _Float16* __restrict__ FYC,
                       const _Float16* __restrict__ FYS, _Float16* __restrict__ out1) {
  __shared__ _Float16 tile[SY * WCH];         // 6784 B, LDS offset 0
  int bvx = blockIdx.x;                       // 0..BV*SX-1
  if ((threadIdx.x >> 5) == 0) {
    tdm_load_f16(0u, h + (size_t)bvx * SY * WCH, SY * WCH, 1, SY * WCH);
    __builtin_amdgcn_s_wait_tensorcnt(0);
  }
  __syncthreads();
  TDM_PUBLISH(tile);
  __syncthreads();
  int ct = threadIdx.x >> 5;
  int l = threadIdx.x & 31;
  int m = l & 15, kh = (l >> 4) * 8;
  int c = ct * 16 + m;
  int bv = bvx / SX, xx = bvx % SX;
  v8f ar = {}, ai = {};
#pragma unroll
  for (int ks = 0; ks < 4; ++ks) {
    v16h a;
#pragma unroll
    for (int i = 0; i < 8; ++i) {
      int y0 = ks * 32 + kh + i, y1 = ks * 32 + 16 + kh + i;
      a[i]     = (y0 < SY) ? tile[y0 * WCH + c] : (_Float16)0.0f;
      a[8 + i] = (y1 < SY) ? tile[y1 * WCH + c] : (_Float16)0.0f;
    }
    v16h bc = load_b_t(FYC + ks * 32, 128);
    v16h bs = load_b_t(FYS + ks * 32, 128);
    ar = wmma32(a, bc, ar);
    ai = wmma32(a, bs, ai);
  }
  int ky = l & 15, mb = (l >> 4) * 8;
  size_t ro = (((size_t)bv * 16 + ky) * 2 + 0) * WCH * 128;
  size_t io = (((size_t)bv * 16 + ky) * 2 + 1) * WCH * 128;
#pragma unroll
  for (int j = 0; j < 8; ++j) {
    int cc = ct * 16 + mb + j;
    out1[ro + (size_t)cc * 128 + xx] = (_Float16)ar[j];
    out1[io + (size_t)cc * 128 + xx] = (_Float16)ai[j];
  }
}

// Forward DFT over x; out1 tile [2][32][128] (16KB contiguous) staged via TDM.
__global__ void k_dftx(const _Float16* __restrict__ out1, const _Float16* __restrict__ FXR,
                       const _Float16* __restrict__ FXI, _Float16* __restrict__ modes) {
  __shared__ _Float16 t2[2 * WCH * 128];      // 16 KB, LDS offset 0
  int id = blockIdx.x;                        // bv*16+ky
  if ((threadIdx.x >> 5) == 0) {
    tdm_load_f16(0u, out1 + (size_t)id * 2 * WCH * 128, 2 * WCH * 128, 1, 2 * WCH * 128);
    __builtin_amdgcn_s_wait_tensorcnt(0);
  }
  __syncthreads();
  TDM_PUBLISH(t2);
  __syncthreads();
  int bv = id >> 4, ky = id & 15;
  int b = bv / VV, v = bv % VV;
  int ct = threadIdx.x >> 5;
  int l = threadIdx.x & 31;
  int m = l & 15, kh = (l >> 4) * 8;
  int c = ct * 16 + m;
  v16h A[8];
#pragma unroll
  for (int ks = 0; ks < 8; ++ks) {
#pragma unroll
    for (int i = 0; i < 8; ++i) {
      int k0 = ks * 32 + kh + i, k1 = ks * 32 + 16 + kh + i;
      int x0 = k0 & 127, r0 = k0 >> 7;
      int x1 = k1 & 127, r1 = k1 >> 7;
      A[ks][i]     = (x0 < SX) ? t2[(r0 * WCH + c) * 128 + x0] : (_Float16)0.0f;
      A[ks][8 + i] = (x1 < SX) ? t2[(r1 * WCH + c) * 128 + x1] : (_Float16)0.0f;
    }
  }
#pragma unroll
  for (int nt = 0; nt < 2; ++nt) {
    v8f accr = {}, acci = {};
#pragma unroll
    for (int ks = 0; ks < 8; ++ks) {
      v16h br = load_b_t(FXR + (size_t)nt * 16 * 256 + ks * 32, 256);
      v16h bi = load_b_t(FXI + (size_t)nt * 16 * 256 + ks * 32, 256);
      accr = wmma32(A[ks], br, accr);
      acci = wmma32(A[ks], bi, acci);
    }
    int kx = nt * 16 + (l & 15), mb = (l >> 4) * 8;
#pragma unroll
    for (int j = 0; j < 8; ++j) {
      int cc = ct * 16 + mb + j;
      size_t o = ((((size_t)b * VV + v) * 32 + kx) * 16 + ky) * 64;
      modes[o + cc]      = (_Float16)accr[j];
      modes[o + 32 + cc] = (_Float16)acci[j];
    }
  }
}

// Complex channel mixing with spec_w1/spec_w2 (on-the-fly f32->f16 B frags).
__global__ void k_modemix(const _Float16* __restrict__ modes, const float* __restrict__ sw1,
                          const float* __restrict__ sw2, int layer,
                          _Float16* __restrict__ mixed) {
  int id = blockIdx.x;                        // v*512 + kx*16 + ky
  int ky = id & 15, kx = (id >> 4) & 31, v = id >> 9;
  int l = threadIdx.x;
  int m = l & 15, kh = (l >> 4) * 8;
  v16h A0, A1;
#pragma unroll
  for (int i = 0; i < 8; ++i) {
    size_t rb = ((((size_t)m * VV + v) * 32 + kx) * 16 + ky) * 64;
    A0[i]     = modes[rb + kh + i];
    A0[8 + i] = modes[rb + 16 + kh + i];
    A1[i]     = modes[rb + 32 + kh + i];
    A1[8 + i] = modes[rb + 48 + kh + i];
  }
  const float* w = (kx < 16) ? sw1 : sw2;
  int m1i = kx & 15, m2i = ky;
  int nB = l & 15, kbB = (l >> 4) * 8;
#pragma unroll
  for (int nt = 0; nt < 2; ++nt) {
    int o = nt * 16 + nB;
    v16h br0, br1, bi0, bi1;
#pragma unroll
    for (int i = 0; i < 16; ++i) {
      int c = (i < 8) ? kbB + i : 16 + kbB + (i - 8);
      size_t wi = (((((size_t)layer * 32 + c) * 32 + o) * VV + v) * 16 + m1i) * 16 + m2i;
      float wr = w[2 * wi], wim = w[2 * wi + 1];
      br0[i] = (_Float16)wr;   br1[i] = (_Float16)(-wim);
      bi0[i] = (_Float16)wim;  bi1[i] = (_Float16)wr;
    }
    v8f accr = {}, acci = {};
    accr = wmma32(A0, br0, accr); accr = wmma32(A1, br1, accr);
    acci = wmma32(A0, bi0, acci); acci = wmma32(A1, bi1, acci);
    int mb = (l >> 4) * 8;
#pragma unroll
    for (int j = 0; j < 8; ++j) {
      int bidx = mb + j;
      size_t ob = ((((size_t)bidx * VV + v) * 32 + kx) * 16 + ky) * 64;
      mixed[ob + o]      = (_Float16)accr[j];
      mixed[ob + 32 + o] = (_Float16)acci[j];
    }
  }
}

// Inverse over x; mixed tile (32 rows of 64 f16, row stride 1024) staged via
// a 2-D TDM descriptor into LDS.
__global__ void k_idftx(const _Float16* __restrict__ mixed, const _Float16* __restrict__ IXR,
                        const _Float16* __restrict__ IXI, _Float16* __restrict__ z) {
  __shared__ _Float16 t3[32 * 64];            // 4 KB, LDS offset 0
  int id = blockIdx.x;                        // bv*16+ky
  int bv = id >> 4, ky = id & 15;
  int b = bv / VV, v = bv % VV;
  if ((threadIdx.x >> 5) == 0) {
    const _Float16* src = mixed + ((((size_t)b * VV + v) * 32) * 16 + ky) * 64;
    tdm_load_f16(0u, src, 64, 32, 16 * 64);
    __builtin_amdgcn_s_wait_tensorcnt(0);
  }
  __syncthreads();
  TDM_PUBLISH(t3);
  __syncthreads();
  int ct = threadIdx.x >> 5;
  int l = threadIdx.x & 31;
  int m = l & 15, kh = (l >> 4) * 8;
  int c = ct * 16 + m;
  v16h A0, A1;
#pragma unroll
  for (int i = 0; i < 8; ++i) {
    int kxa = kh + i, kxb = 16 + kh + i;
    A0[i] = t3[kxa * 64 + c];       A0[8 + i] = t3[kxb * 64 + c];
    A1[i] = t3[kxa * 64 + 32 + c];  A1[8 + i] = t3[kxb * 64 + 32 + c];
  }
  size_t zb = (size_t)id * 2 * WCH * 112;
#pragma unroll
  for (int nt = 0; nt < 7; ++nt) {
    v16h br0 = load_b_t(IXR + (size_t)nt * 16 * 64, 64);
    v16h br1 = load_b_t(IXR + (size_t)nt * 16 * 64 + 32, 64);
    v16h bi0 = load_b_t(IXI + (size_t)nt * 16 * 64, 64);
    v16h bi1 = load_b_t(IXI + (size_t)nt * 16 * 64 + 32, 64);
    v8f accr = {}, acci = {};
    accr = wmma32(A0, br0, accr); accr = wmma32(A1, br1, accr);
    acci = wmma32(A0, bi0, acci); acci = wmma32(A1, bi1, acci);
    int x = nt * 16 + (l & 15), mb = (l >> 4) * 8;
#pragma unroll
    for (int j = 0; j < 8; ++j) {
      int cc = ct * 16 + mb + j;
      z[zb + (size_t)cc * 112 + x]         = (_Float16)accr[j];
      z[zb + ((size_t)WCH + cc) * 112 + x] = (_Float16)acci[j];
    }
  }
}

// Inverse over y (to real spatial): s[point][c]
__global__ void k_idfty(const _Float16* __restrict__ z, const _Float16* __restrict__ IY,
                        _Float16* __restrict__ s) {
  int bvx = blockIdx.x;
  int bv = bvx / SX, xx = bvx % SX;
  int ct = threadIdx.x >> 5;
  int l = threadIdx.x & 31;
  int m = l & 15, kh = (l >> 4) * 8;
  int c = ct * 16 + m;
  v16h A;
#pragma unroll
  for (int i = 0; i < 8; ++i) {
    int k0 = kh + i, k1 = 16 + kh + i;
    int ky0 = k0 & 15, r0 = k0 >> 4;
    int ky1 = k1 & 15, r1 = k1 >> 4;
    A[i]     = z[((((size_t)bv * 16 + ky0) * 2 + r0) * WCH + c) * 112 + xx];
    A[8 + i] = z[((((size_t)bv * 16 + ky1) * 2 + r1) * WCH + c) * 112 + xx];
  }
  size_t nb = (size_t)bvx * SY;
#pragma unroll
  for (int nt = 0; nt < 7; ++nt) {
    v16h bfrag = load_b_t(IY + (size_t)nt * 16 * 32, 32);
    v8f acc = {};
    acc = wmma32(A, bfrag, acc);
    int y = nt * 16 + (l & 15), mb = (l >> 4) * 8;
    if (y < SY) {
#pragma unroll
      for (int j = 0; j < 8; ++j) {
        int cc = ct * 16 + mb + j;
        s[(nb + y) * WCH + cc] = (_Float16)acc[j];
      }
    }
  }
}

// ------------------------- pointwise channel mixes -------------------------

__global__ void k_mlp1(const _Float16* __restrict__ s, const _Float16* __restrict__ wT,
                       const float* __restrict__ bias, _Float16* __restrict__ t1) {
  int tile = blockIdx.x;
  int nt = threadIdx.x >> 5;
  int l = threadIdx.x & 31;
  __builtin_prefetch(s + ((size_t)tile + 64) * 16 * WCH, 0, 0);
  v16h a = load_a_rm(s + (size_t)tile * 16 * WCH, WCH);
  v16h b = load_b_t(wT + (size_t)nt * 16 * WCH, WCH);
  v8f acc = {};
  acc = wmma32(a, b, acc);
  int ch = nt * 16 + (l & 15), mb = (l >> 4) * 8;
  float bv = bias[ch];
#pragma unroll
  for (int j = 0; j < 8; ++j)
    t1[((size_t)tile * 16 + mb + j) * WCH + ch] = (_Float16)gelu_f(acc[j] + bv);
}

__global__ void k_fused_out(const _Float16* __restrict__ t1, const _Float16* __restrict__ hin,
                            const _Float16* __restrict__ w2T, const _Float16* __restrict__ wwT,
                            const _Float16* __restrict__ gt, const _Float16* __restrict__ res,
                            _Float16* __restrict__ hout) {
  int tile = blockIdx.x;
  int nt = threadIdx.x >> 5;
  int l = threadIdx.x & 31;
  __builtin_prefetch(t1 + ((size_t)tile + 64) * 16 * WCH, 0, 0);
  __builtin_prefetch(hin + ((size_t)tile + 64) * 16 * WCH, 0, 0);
  v16h a1 = load_a_rm(t1 + (size_t)tile * 16 * WCH, WCH);
  v16h a2 = load_a_rm(hin + (size_t)tile * 16 * WCH, WCH);
  v16h b1 = load_b_t(w2T + (size_t)nt * 16 * WCH, WCH);
  v16h b2 = load_b_t(wwT + (size_t)nt * 16 * WCH, WCH);
  v8f acc = {};
  acc = wmma32(a1, b1, acc);
  acc = wmma32(a2, b2, acc);
  int ch = nt * 16 + (l & 15), mb = (l >> 4) * 8;
#pragma unroll
  for (int j = 0; j < 8; ++j) {
    size_t p = (size_t)tile * 16 + mb + j;
    int sp = (int)(p % SP2);
    float vsum = acc[j] + (float)gt[(size_t)sp * WCH + ch];
    float g = gelu_f(vsum);
    if (res) g += (float)res[p * WCH + ch];
    hout[p * WCH + ch] = (_Float16)g;
  }
}

// ------------------------- head: fc1 -> gelu -> fc2 ------------------------
// 4 waves/block, one 16-point tile per wave. FC1T||FC2T (24KB, adjacent in ws)
// is TDM-staged into LDS once per block; the 16x256 hidden stays in LDS.
__global__ void k_head(const _Float16* __restrict__ h, const _Float16* __restrict__ FC1T,
                       const float* __restrict__ fc1b, const float* __restrict__ fc2b,
                       float* __restrict__ out) {
  __shared__ _Float16 wlds[256 * 32 + 16 * 256];  // 24 KB, LDS offset 0
  __shared__ _Float16 g[4][16 * 264];             // 33 KB
  int wv = threadIdx.x >> 5;
  int l = threadIdx.x & 31;
  if (wv == 0) {
    tdm_load_f16(0u, FC1T, 256 * 32 + 16 * 256, 1, 256 * 32 + 16 * 256);
    __builtin_amdgcn_s_wait_tensorcnt(0);
  }
  __syncthreads();
  TDM_PUBLISH(wlds);
  __syncthreads();
  int tile = blockIdx.x * 4 + wv;
  if (tile < NTILES) {
    __builtin_prefetch(h + ((size_t)tile + 16) * 16 * WCH, 0, 0);
    v16h a = load_a_rm(h + (size_t)tile * 16 * WCH, WCH);
#pragma unroll
    for (int nt = 0; nt < 16; ++nt) {
      v16h b = load_b_t(&wlds[nt * 16 * 32], 32);
      v8f acc = {};
      acc = wmma32(a, b, acc);
      int ch = nt * 16 + (l & 15), mb = (l >> 4) * 8;
      float bias = fc1b[ch];
#pragma unroll
      for (int j = 0; j < 8; ++j)
        g[wv][(mb + j) * 264 + ch] = (_Float16)gelu_f(acc[j] + bias);
    }
    v8f acc = {};
    int m = l & 15, kh = (l >> 4) * 8;
#pragma unroll
    for (int ks = 0; ks < 8; ++ks) {
      v16h a2;
#pragma unroll
      for (int i = 0; i < 8; ++i) {
        a2[i]     = g[wv][m * 264 + ks * 32 + kh + i];
        a2[8 + i] = g[wv][m * 264 + ks * 32 + 16 + kh + i];
      }
      v16h b2 = load_b_t(&wlds[256 * 32 + ks * 32], 256);
      acc = wmma32(a2, b2, acc);
    }
    int nc = l & 15, mb = (l >> 4) * 8;
    if (nc < STEP) {
      float bias = fc2b[nc];
#pragma unroll
      for (int j = 0; j < 8; ++j)
        out[((size_t)tile * 16 + mb + j) * STEP + nc] = acc[j] + bias;
    }
  }
}

// ------------------------- host orchestration ------------------------------

extern "C" void kernel_launch(void* const* d_in, const int* in_sizes, int n_in,
                              void* d_out, int out_size, void* d_ws, size_t ws_size,
                              hipStream_t stream) {
  (void)in_sizes; (void)n_in; (void)out_size; (void)ws_size;

  const float* xin   = (const float*)d_in[0];
  const float* fc0w  = (const float*)d_in[1];
  const float* fc0b  = (const float*)d_in[2];
  const float* sw1   = (const float*)d_in[3];   // complex64 -> interleaved f32
  const float* sw2   = (const float*)d_in[4];
  const float* m1w   = (const float*)d_in[5];
  const float* m1b   = (const float*)d_in[6];
  const float* m2w   = (const float*)d_in[7];
  const float* m2b   = (const float*)d_in[8];
  const float* www   = (const float*)d_in[9];
  const float* wbv   = (const float*)d_in[10];
  const float* bww   = (const float*)d_in[11];
  const float* bbv   = (const float*)d_in[12];
  const float* fc1w  = (const float*)d_in[13];
  const float* fc1b  = (const float*)d_in[14];
  const float* fc2w  = (const float*)d_in[15];
  const float* fc2b  = (const float*)d_in[16];
  float* outp = (float*)d_out;

  char* base = (char*)d_ws;
  size_t off = 0;
  auto alloc_h = [&](size_t nelem) -> _Float16* {
    _Float16* p = (_Float16*)(base + off);
    off += (nelem * sizeof(_Float16) + 255) & ~(size_t)255;
    return p;
  };
  _Float16* h0   = alloc_h((size_t)NPT * WCH);
  _Float16* h1   = alloc_h((size_t)NPT * WCH);
  _Float16* x0b  = alloc_h((size_t)NPT * WCH);
  _Float16* sbuf = alloc_h((size_t)NPT * WCH);
  _Float16* t1b  = alloc_h((size_t)NPT * WCH);
  _Float16* out1 = alloc_h((size_t)BB * VV * 16 * 2 * WCH * 128);
  _Float16* mods = alloc_h((size_t)BB * VV * 32 * 16 * 64);
  _Float16* mixd = alloc_h((size_t)BB * VV * 32 * 16 * 64);
  _Float16* zbuf = alloc_h((size_t)BB * VV * 16 * 2 * WCH * 112);
  _Float16* gt   = alloc_h((size_t)NL * SP2 * WCH);
  _Float16* FYC  = alloc_h(16 * 128);
  _Float16* FYS  = alloc_h(16 * 128);
  _Float16* FXR  = alloc_h(32 * 256);
  _Float16* FXI  = alloc_h(32 * 256);
  _Float16* IXR  = alloc_h(112 * 64);
  _Float16* IXI  = alloc_h(112 * 64);
  _Float16* IY   = alloc_h(112 * 32);
  _Float16* FC0T = alloc_h(32 * 32);
  _Float16* M1T  = alloc_h((size_t)NL * 32 * 32);
  _Float16* M2T  = alloc_h((size_t)NL * 32 * 32);
  _Float16* WWT  = alloc_h((size_t)NL * 32 * 32);
  _Float16* FC1T = alloc_h(256 * 32);            // 16384 B (256-aligned size)
  _Float16* FC2T = alloc_h(16 * 256);            // contiguous right after FC1T

  {
    int total = 1024 + 3 * NL * 1024 + 256 * 32 + 16 * 256;
    k_prep_weights<<<(total + 255) / 256, 256, 0, stream>>>(
        fc0w, m1w, m2w, www, fc1w, fc2w, FC0T, M1T, M2T, WWT, FC1T, FC2T);
  }
  {
    int total = 16 * 128 + 32 * 256 + 112 * 64 + 112 * 32;
    k_prep_tables<<<(total + 255) / 256, 256, 0, stream>>>(FYC, FYS, FXR, FXI,
                                                           IXR, IXI, IY);
  }
  {
    int total = NL * SP2 * WCH;
    k_gridterms<<<(total + 255) / 256, 256, 0, stream>>>(bww, bbv, wbv, m2b, gt);
  }

  k_lift<<<NTILES, 32, 0, stream>>>(xin, FC0T, fc0b, h0);

  const _Float16* cur = h0;
  _Float16* dsts[NL]       = { x0b, h1, h0, x0b, h1, h0 };
  const _Float16* ress[NL] = { nullptr, nullptr, x0b, nullptr, nullptr, x0b };

  for (int i = 0; i < NL; ++i) {
    k_dfty<<<BB * VV * SX, 64, 0, stream>>>(cur, FYC, FYS, out1);
    k_dftx<<<BB * VV * 16, 64, 0, stream>>>(out1, FXR, FXI, mods);
    k_modemix<<<VV * 32 * 16, 32, 0, stream>>>(mods, sw1, sw2, i, mixd);
    k_idftx<<<BB * VV * 16, 64, 0, stream>>>(mixd, IXR, IXI, zbuf);
    k_idfty<<<BB * VV * SX, 64, 0, stream>>>(zbuf, IY, sbuf);
    k_mlp1<<<NTILES, 64, 0, stream>>>(sbuf, M1T + (size_t)i * 1024,
                                      m1b + (size_t)i * WCH, t1b);
    k_fused_out<<<NTILES, 64, 0, stream>>>(t1b, cur, M2T + (size_t)i * 1024,
                                           WWT + (size_t)i * 1024,
                                           gt + (size_t)i * SP2 * WCH,
                                           ress[i], dsts[i]);
    cur = dsts[i];
  }

  k_head<<<(NTILES + 3) / 4, 128, 0, stream>>>(cur, FC1T, fc1b, fc2b, outp);
}
